// MultiScaleWindowAttention_45423574122993
// MI455X (gfx1250) — compile-verified, matrix operand order checked by reading, not run on hardware
//
#include <hip/hip_runtime.h>

// ---------------- problem constants ----------------
#define BB     128
#define NN     343
#define CC     384
#define HH     12
#define DD     32
#define NWIN   64
#define NSCALE 3
#define KTOT   384               // C (contraction dim for all dense GEMMs)
#define MROWS  (BB * NN)         // 43904, divisible by 16
#define MT     (MROWS / 16)      // 2744 m-tiles
#define SCALE_F 0.1767766952966369f   // 32^-0.5

typedef __attribute__((ext_vector_type(16))) __bf16 v16bf;
typedef __attribute__((ext_vector_type(8)))  float  v8f;

// ---------------- bf16 helpers (manual RNE, no __bf16 arithmetic) ----------
__device__ __forceinline__ unsigned short f2bf(float f) {
    unsigned int u = __float_as_uint(f);
    unsigned int r = u + 0x7FFFu + ((u >> 16) & 1u);
    return (unsigned short)(r >> 16);
}

// ---------------- WMMA wrapper ----------------
__device__ __forceinline__ v8f wmma_bf(v16bf a, v16bf b, v8f c) {
    return __builtin_amdgcn_wmma_f32_16x16x32_bf16(
        /*neg_a=*/false, a, /*neg_b=*/false, b,
        /*c_mod=*/(short)0, c, /*reuse_a=*/false, /*reuse_b=*/false);
}

// Fragment loader for a K-contiguous row (bf16), following ISA 16-bit A/B layout:
// lanes 0-15 hold K = {0..7, 16..23}, lanes 16-31 hold K = {8..15, 24..31}.
__device__ __forceinline__ v16bf load_row_frag(const unsigned short* row, int lane) {
    int off = (lane & 16) ? 8 : 0;
    union { uint4 u4[2]; v16bf v; } u;
    u.u4[0] = *(const uint4*)(row + off);        // K = off .. off+7
    u.u4[1] = *(const uint4*)(row + off + 16);   // K = off+16 .. off+23
    return u.v;
}

// Same fragment but converting from an f32 row on the fly.
__device__ __forceinline__ v16bf load_row_frag_f32(const float* row, int lane) {
    int off = (lane & 16) ? 8 : 0;
    float4 a = *(const float4*)(row + off);
    float4 b = *(const float4*)(row + off + 4);
    float4 c = *(const float4*)(row + off + 16);
    float4 d = *(const float4*)(row + off + 20);
    union { unsigned short s[16]; v16bf v; } u;
    u.s[0]  = f2bf(a.x); u.s[1]  = f2bf(a.y); u.s[2]  = f2bf(a.z); u.s[3]  = f2bf(a.w);
    u.s[4]  = f2bf(b.x); u.s[5]  = f2bf(b.y); u.s[6]  = f2bf(b.z); u.s[7]  = f2bf(b.w);
    u.s[8]  = f2bf(c.x); u.s[9]  = f2bf(c.y); u.s[10] = f2bf(c.z); u.s[11] = f2bf(c.w);
    u.s[12] = f2bf(d.x); u.s[13] = f2bf(d.y); u.s[14] = f2bf(d.z); u.s[15] = f2bf(d.w);
    return u.v;
}

// Gather a B-fragment of V for the P x V product: contract dim = attention
// column c (rows of V), N dim = head dim d (fixed per lane).
__device__ __forceinline__ v16bf gather_v_frag(const unsigned short* V, int cbase,
                                               int d, int lane) {
    union { unsigned short s[16]; v16bf v; } u;
    int g = (lane & 16) ? 8 : 0;
#pragma unroll
    for (int t = 0; t < 16; ++t) {
        int kk = (t < 8 ? t : t + 8) + g;
        int r  = cbase + kk;
        if (r > NN - 1) r = NN - 1;          // p==0 there, address-safe clamp
        u.s[t] = V[(size_t)r * DD + d];
    }
    return u.v;
}

// Shared 16x16-tile GEMM core: D(mtile,ntile) = A[M,K] * W[N,K]^T  (bf16, f32 acc)
__device__ __forceinline__ v8f gemm_tile(const unsigned short* __restrict__ A,
                                         const unsigned short* __restrict__ W,
                                         int mtile, int ntile) {
    int lane = threadIdx.x & 31;
    int nl   = lane & 15;
    const unsigned short* arow = A + (size_t)(mtile * 16 + nl) * KTOT;
    const unsigned short* wrow = W + (size_t)(ntile * 16 + nl) * KTOT;
    v8f acc = {};
    for (int kc = 0; kc < KTOT; kc += 32) {
        v16bf a = load_row_frag(arow + kc, lane);
        v16bf b = load_row_frag(wrow + kc, lane);
        acc = wmma_bf(a, b, acc);
    }
    return acc;
}

// ---------------- prep kernels ----------------
__global__ void fw_softmax_kernel(const float* __restrict__ fusion_w,
                                  float* __restrict__ fw) {
    if (threadIdx.x == 0) {
        float a = fusion_w[0], b = fusion_w[1], c = fusion_w[2];
        float m = fmaxf(a, fmaxf(b, c));
        float e0 = __expf(a - m), e1 = __expf(b - m), e2 = __expf(c - m);
        float s = e0 + e1 + e2;
        fw[0] = e0 / s; fw[1] = e1 / s; fw[2] = e2 / s;
    }
}

__global__ void cvt_x_kernel(const float* __restrict__ x,
                             unsigned short* __restrict__ xbf,
                             float* __restrict__ fused, size_t n) {
    size_t i = (size_t)blockIdx.x * blockDim.x + threadIdx.x;
    if (i < n) { xbf[i] = f2bf(x[i]); fused[i] = 0.0f; }
}

__global__ void cvt_kernel(const float* __restrict__ src,
                           unsigned short* __restrict__ dst, size_t n) {
    size_t i = (size_t)blockIdx.x * blockDim.x + threadIdx.x;
    if (i < n) dst[i] = f2bf(src[i]);
}

// ---------------- stage 1: QKV GEMM (one scale) ----------------
// y = x @ qkv_w^T, scattered into [which(q/k/v)][B][H][N][D] bf16, q pre-scaled.
__global__ __launch_bounds__(256) void qkv_gemm_kernel(
    const unsigned short* __restrict__ xbf,      // [MROWS][C] bf16
    const unsigned short* __restrict__ wq,       // [1152][384] bf16 (this scale)
    unsigned short* __restrict__ qkvout) {       // [3][B][H][N][D] bf16
    int wave  = threadIdx.x >> 5;
    int lane  = threadIdx.x & 31;
    int nl    = lane & 15;
    int g8    = (lane & 16) ? 8 : 0;
    int mtile = blockIdx.x;
    int ntile = blockIdx.y * 8 + wave;           // 0..71  (N = 1152)
    v8f acc = gemm_tile(xbf, wq, mtile, ntile);
#pragma unroll
    for (int e = 0; e < 8; ++e) {
        int mrow = mtile * 16 + e + g8;
        int b = mrow / NN;
        int n = mrow - b * NN;
        int o = ntile * 16 + nl;
        int which = o / CC;
        int rem   = o - which * CC;
        int h = rem / DD;
        int d = rem - h * DD;
        float val = acc[e];
        if (which == 0) val *= SCALE_F;          // pre-scale Q
        size_t idx = (((size_t)(which * BB + b) * HH + h) * NN + n) * DD + d;
        qkvout[idx] = f2bf(val);
    }
}

// ---------------- stage 2: windowed flash attention (one scale) ----------
__global__ __launch_bounds__(128) void attn_kernel(
    const unsigned short* __restrict__ qkv,      // [3][B][H][N][D] bf16
    const float* __restrict__ mask,              // [NW][N][N] f32
    unsigned short* __restrict__ attnout) {      // [B][N][C] bf16
    __shared__ unsigned short sP[4 * 16 * 32];   // per-wave P staging tiles
    int wave  = threadIdx.x >> 5;
    int lane  = threadIdx.x & 31;
    int nl    = lane & 15;
    int g8    = (lane & 16) ? 8 : 0;
    int mtile = blockIdx.x * 4 + wave;           // 22 row-tiles of 16
    if (mtile >= 22) return;                      // wave-uniform exit
    int h = blockIdx.y;
    int b = blockIdx.z;
    int w = b % NWIN;

    const unsigned short* Q  = qkv + ((size_t)(0 * BB + b) * HH + h) * NN * DD;
    const unsigned short* Kp = qkv + ((size_t)(1 * BB + b) * HH + h) * NN * DD;
    const unsigned short* Vp = qkv + ((size_t)(2 * BB + b) * HH + h) * NN * DD;
    const float* Mw = mask + (size_t)w * NN * NN;
    unsigned short* sPw = sP + wave * 512;

    int mbase = mtile * 16;
    int qrow  = mbase + nl; if (qrow > NN - 1) qrow = NN - 1;
    v16bf qA = load_row_frag(Q + (size_t)qrow * DD, lane);

    v8f acc0 = {}, acc1 = {};
    float rm[8], rl[8];
#pragma unroll
    for (int e = 0; e < 8; ++e) { rm[e] = -3.0e38f; rl[e] = 0.0f; }

    for (int j = 0; j < 11; ++j) {               // 11 chunks of 32 cols (352>=343)
        int cbase = j * 32;
        int c0 = cbase + nl;
        int c1 = c0 + 16;
        int kr0 = c0 > NN - 1 ? NN - 1 : c0;
        int kr1 = c1 > NN - 1 ? NN - 1 : c1;
        v16bf kB0 = load_row_frag(Kp + (size_t)kr0 * DD, lane);
        v16bf kB1 = load_row_frag(Kp + (size_t)kr1 * DD, lane);
        v8f z = {};
        v8f s0 = wmma_bf(qA, kB0, z);
        v8f s1 = wmma_bf(qA, kB1, z);
#pragma unroll
        for (int e = 0; e < 8; ++e) {
            int m  = mbase + e + g8;
            int mc = m > NN - 1 ? NN - 1 : m;
            const float* mrow = Mw + (size_t)mc * NN;
            float v0 = (c0 < NN) ? s0[e] + mrow[c0] : -3.0e38f;
            float v1 = (c1 < NN) ? s1[e] + mrow[c1] : -3.0e38f;
            float t = fmaxf(v0, v1);
#pragma unroll
            for (int msk = 1; msk < 16; msk <<= 1) t = fmaxf(t, __shfl_xor(t, msk, 32));
            float mnew = fmaxf(rm[e], t);
            float f = __expf(rm[e] - mnew);       // 0 on first valid chunk
            rm[e] = mnew;
            rl[e] *= f;
            acc0[e] *= f;
            acc1[e] *= f;
            float p0 = __expf(v0 - mnew);
            float p1 = __expf(v1 - mnew);
            float ps = p0 + p1;
#pragma unroll
            for (int msk = 1; msk < 16; msk <<= 1) ps += __shfl_xor(ps, msk, 32);
            rl[e] += ps;
            int ml = e + g8;
            sPw[ml * 32 + nl]      = f2bf(p0);    // P tile row-major in LDS
            sPw[ml * 32 + 16 + nl] = f2bf(p1);
        }
        // Re-read P as an A-fragment (same-wave DS ordering; no barrier needed)
        union { uint4 u4[2]; v16bf v; } up;
        up.u4[0] = *(const uint4*)(sPw + nl * 32 + g8);
        up.u4[1] = *(const uint4*)(sPw + nl * 32 + g8 + 16);
        v16bf pA  = up.v;
        v16bf vB0 = gather_v_frag(Vp, cbase, nl,      lane);   // d = 0..15
        v16bf vB1 = gather_v_frag(Vp, cbase, nl + 16, lane);   // d = 16..31
        acc0 = wmma_bf(pA, vB0, acc0);
        acc1 = wmma_bf(pA, vB1, acc1);
    }
#pragma unroll
    for (int e = 0; e < 8; ++e) {
        int m = mbase + e + g8;
        if (m < NN) {
            float inv = 1.0f / rl[e];
            size_t base = ((size_t)b * NN + m) * CC + h * DD;
            attnout[base + nl]      = f2bf(acc0[e] * inv);
            attnout[base + 16 + nl] = f2bf(acc1[e] * inv);
        }
    }
}

// ---------------- stage 3: proj GEMM + weighted fusion (one scale) --------
__global__ __launch_bounds__(256) void proj_gemm_kernel(
    const unsigned short* __restrict__ a,        // attnout [MROWS][C] bf16
    const unsigned short* __restrict__ wp,       // [384][384] bf16 (this scale)
    const float* __restrict__ pb,                // [384] f32 (this scale)
    const float* __restrict__ fw, int s,
    float* __restrict__ fused) {                 // [MROWS][C] f32 accumulator
    int wave  = threadIdx.x >> 5;
    int lane  = threadIdx.x & 31;
    int nl    = lane & 15;
    int g8    = (lane & 16) ? 8 : 0;
    int mtile = blockIdx.x;
    int ntile = blockIdx.y * 8 + wave;           // 0..23
    v8f acc = gemm_tile(a, wp, mtile, ntile);
    float f = fw[s];
    int o = ntile * 16 + nl;
    float bias = pb[o];
#pragma unroll
    for (int e = 0; e < 8; ++e) {
        int mrow = mtile * 16 + e + g8;
        fused[(size_t)mrow * CC + o] += f * (acc[e] + bias);
    }
}

// ---------------- stage 4: final out projection ----------------
__global__ __launch_bounds__(256) void out_gemm_kernel(
    const float* __restrict__ fusedA,            // [MROWS][C] f32
    const unsigned short* __restrict__ wo,       // [384][384] bf16
    const float* __restrict__ ob,                // [384] f32
    float* __restrict__ out) {                   // [MROWS][C] f32
    int wave  = threadIdx.x >> 5;
    int lane  = threadIdx.x & 31;
    int nl    = lane & 15;
    int g8    = (lane & 16) ? 8 : 0;
    int mtile = blockIdx.x;
    int ntile = blockIdx.y * 8 + wave;
    const float* arow = fusedA + (size_t)(mtile * 16 + nl) * KTOT;
    const unsigned short* wrow = wo + (size_t)(ntile * 16 + nl) * KTOT;
    v8f acc = {};
    for (int kc = 0; kc < KTOT; kc += 32) {
        v16bf a = load_row_frag_f32(arow + kc, lane);
        v16bf b = load_row_frag(wrow + kc, lane);
        acc = wmma_bf(a, b, acc);
    }
    int o = ntile * 16 + nl;
    float bias = ob[o];
#pragma unroll
    for (int e = 0; e < 8; ++e) {
        int mrow = mtile * 16 + e + g8;
        out[(size_t)mrow * CC + o] = acc[e] + bias;
    }
}

// ---------------- host side ----------------
extern "C" void kernel_launch(void* const* d_in, const int* in_sizes, int n_in,
                              void* d_out, int out_size, void* d_ws, size_t ws_size,
                              hipStream_t stream) {
    (void)in_sizes; (void)n_in; (void)out_size; (void)ws_size;
    const float* x        = (const float*)d_in[0];
    const float* mask     = (const float*)d_in[1];
    const float* qkv_w    = (const float*)d_in[2];
    const float* proj_w   = (const float*)d_in[3];
    const float* proj_b   = (const float*)d_in[4];
    const float* fusion_w = (const float*)d_in[5];
    const float* out_w    = (const float*)d_in[6];
    const float* out_b    = (const float*)d_in[7];
    float* out = (float*)d_out;

    char* ws = (char*)d_ws;
    size_t off = 0;
    auto wsalloc = [&](size_t bytes) -> char* {
        char* p = ws + off;
        off += (bytes + 255) & ~(size_t)255;
        return p;
    };
    float*          fw      = (float*)wsalloc(3 * sizeof(float));
    unsigned short* xbf     = (unsigned short*)wsalloc((size_t)MROWS * CC * 2);
    unsigned short* wq      = (unsigned short*)wsalloc((size_t)NSCALE * 3 * CC * CC * 2);
    unsigned short* wp      = (unsigned short*)wsalloc((size_t)NSCALE * CC * CC * 2);
    unsigned short* wo      = (unsigned short*)wsalloc((size_t)CC * CC * 2);
    unsigned short* qkvbuf  = (unsigned short*)wsalloc((size_t)3 * MROWS * CC * 2);
    unsigned short* attnout = (unsigned short*)wsalloc((size_t)MROWS * CC * 2);
    float*          fused   = (float*)wsalloc((size_t)MROWS * CC * 4);

    fw_softmax_kernel<<<1, 32, 0, stream>>>(fusion_w, fw);
    size_t nx = (size_t)MROWS * CC;
    cvt_x_kernel<<<(unsigned)((nx + 255) / 256), 256, 0, stream>>>(x, xbf, fused, nx);
    size_t nq = (size_t)NSCALE * 3 * CC * CC;
    cvt_kernel<<<(unsigned)((nq + 255) / 256), 256, 0, stream>>>(qkv_w, wq, nq);
    size_t np = (size_t)NSCALE * CC * CC;
    cvt_kernel<<<(unsigned)((np + 255) / 256), 256, 0, stream>>>(proj_w, wp, np);
    size_t no = (size_t)CC * CC;
    cvt_kernel<<<(unsigned)((no + 255) / 256), 256, 0, stream>>>(out_w, wo, no);

    for (int s = 0; s < NSCALE; ++s) {
        qkv_gemm_kernel<<<dim3(MT, 9), 256, 0, stream>>>(
            xbf, wq + (size_t)s * 3 * CC * CC, qkvbuf);
        attn_kernel<<<dim3(6, HH, BB), 128, 0, stream>>>(qkvbuf, mask, attnout);
        proj_gemm_kernel<<<dim3(MT, 3), 256, 0, stream>>>(
            attnout, wp + (size_t)s * CC * CC, proj_b + s * CC, fw, s, fused);
    }
    out_gemm_kernel<<<dim3(MT, 3), 256, 0, stream>>>(fused, wo, out_b, out);
}
